// TokenMixingKAN_19825569038553
// MI455X (gfx1250) — compile-verified, matrix operand order checked by reading, not run on hardware
//
#include <hip/hip_runtime.h>
#include <hip/hip_bf16.h>
#include <math.h>

// ---- shapes (fixed by the reference) ----
#define B_DIM 64
#define T_DIM 196
#define C_DIM 768
#define H_DIM 384
#define NROWS (B_DIM * C_DIM)   // 49152 rows for the KAN GEMMs

typedef __bf16 bf16_t;
typedef __attribute__((ext_vector_type(16))) __bf16 v16bf;
typedef __attribute__((ext_vector_type(8)))  __bf16 v8bf;
typedef __attribute__((ext_vector_type(8)))  float  v8f;
typedef __attribute__((ext_vector_type(4)))  float  v4f;

__device__ __forceinline__ bf16_t tobf(float f) {
  union { float f; unsigned u; } v; v.f = f;
  unsigned r = v.u + 0x7FFFu + ((v.u >> 16) & 1u);   // round-to-nearest-even
  unsigned short h = (unsigned short)(r >> 16);
  bf16_t o; __builtin_memcpy(&o, &h, 2);
  return o;
}

// Expand one scalar into [silu(x), B0..B7].  Uniform-knot closed form of the
// cubic Cox-de Boor recursion: knots t_j = -2.2 + 0.4*j (j=0..11); inside
// cell j0 only bases j0-3..j0 are nonzero with the standard uniform cubic
// segment polynomials.  Identical values, ~2.5x fewer VALU ops.
__device__ __forceinline__ void expand9(float x, bf16_t* o) {
  o[0] = tobf(x / (1.0f + __expf(-x)));              // silu
  float u  = (x + 2.2f) * 2.5f;                      // (x - t0) / h
  float fj = floorf(u);
  int   j0 = (int)fj;
  float t  = u - fj;
  float t2 = t * t, t3 = t2 * t;
  const float s = 1.0f / 6.0f;
  float w0 = s * (1.0f - 3.0f * t + 3.0f * t2 - t3); // (1-t)^3/6   (d = 3)
  float w1 = s * (3.0f * t3 - 6.0f * t2 + 4.0f);     //             (d = 2)
  float w2 = s * (-3.0f * t3 + 3.0f * t2 + 3.0f * t + 1.0f); //     (d = 1)
  float w3 = s * t3;                                 //             (d = 0)
  bool inr = (u >= 0.0f) && (j0 <= 10);              // x in [t0, t11)
#pragma unroll
  for (int c = 0; c < 8; ++c) {
    int d = j0 - c;
    float b = (d == 3) ? w0 : (d == 2) ? w1 : (d == 1) ? w2 : (d == 0) ? w3 : 0.0f;
    o[1 + c] = tobf(inr ? b : 0.0f);
  }
}

// ---------------- LayerNorm statistics, two-stage ---------------------------
__global__ void ln_partial_kernel(const float* __restrict__ X,
                                  float* __restrict__ partial) {
  const int b = blockIdx.y;
  const int seg = blockIdx.x;                 // 0..127
  const int n = T_DIM * C_DIM;                // 150528
  const int per = (n + 127) / 128;            // 1176
  const int lo = seg * per;
  const int hi = (lo + per < n) ? lo + per : n;
  const float* xb = X + (size_t)b * n;
  float s = 0.f, q = 0.f;
  for (int i = lo + threadIdx.x; i < hi; i += 256) {
    float v = xb[i]; s += v; q += v * v;
  }
  __shared__ float ss[256], sq[256];
  ss[threadIdx.x] = s; sq[threadIdx.x] = q;
  __syncthreads();
  for (int off = 128; off > 0; off >>= 1) {
    if (threadIdx.x < off) {
      ss[threadIdx.x] += ss[threadIdx.x + off];
      sq[threadIdx.x] += sq[threadIdx.x + off];
    }
    __syncthreads();
  }
  if (threadIdx.x == 0) {
    partial[(b * 128 + seg) * 2 + 0] = ss[0];
    partial[(b * 128 + seg) * 2 + 1] = sq[0];
  }
}

__global__ void ln_final_kernel(const float* __restrict__ partial,
                                float* __restrict__ stats) {
  const int b = blockIdx.x;
  __shared__ float ss[128], sq[128];
  ss[threadIdx.x] = partial[(b * 128 + threadIdx.x) * 2 + 0];
  sq[threadIdx.x] = partial[(b * 128 + threadIdx.x) * 2 + 1];
  __syncthreads();
  for (int off = 64; off > 0; off >>= 1) {
    if (threadIdx.x < off) {
      ss[threadIdx.x] += ss[threadIdx.x + off];
      sq[threadIdx.x] += sq[threadIdx.x + off];
    }
    __syncthreads();
  }
  if (threadIdx.x == 0) {
    const float n = (float)(T_DIM * C_DIM);
    float mean = ss[0] / n;
    float var  = sq[0] / n - mean * mean;
    stats[2 * b]     = mean;
    stats[2 * b + 1] = rsqrtf(var + 1e-5f);
  }
}

// ------------- normalize + (B,T,C) -> (B*C, T) tiled transpose --------------
__global__ void norm_transpose_kernel(const float* __restrict__ X,
                                      const float* __restrict__ lnw,
                                      const float* __restrict__ lnb,
                                      const float* __restrict__ stats,
                                      float* __restrict__ z0) {
  __shared__ float tile[32][33];
  const int b = blockIdx.z;
  const int tBase = blockIdx.y * 32;
  const int cBase = blockIdx.x * 32;
  const float mu = stats[2 * b], rstd = stats[2 * b + 1];
  const int lane = threadIdx.x & 31, grp = threadIdx.x >> 5;
#pragma unroll
  for (int q = 0; q < 4; ++q) {
    int tl = grp + 8 * q, cl = lane;
    int t = tBase + tl, c = cBase + cl;
    float v = 0.f;
    if (t < T_DIM) {
      size_t gi = ((size_t)b * T_DIM + t) * C_DIM + c;
      size_t wi = (size_t)t * C_DIM + c;
      v = (X[gi] - mu) * rstd * lnw[wi] + lnb[wi];
    }
    tile[tl][cl] = v;
  }
  __syncthreads();
#pragma unroll
  for (int q = 0; q < 4; ++q) {
    int cl = grp + 8 * q, tl = lane;
    int t = tBase + tl, c = cBase + cl;
    if (t < T_DIM)
      z0[((size_t)b * C_DIM + c) * T_DIM + t] = tile[tl][cl];
  }
}

// ------------- one-time weight expansion into fragment-native layout --------
// Wexp layout: for 16-col tile ct and K-octet k8 (K index / 8), the 16 lanes'
// 8 bf16 values are contiguous: Wexp[((ct*KT8 + k8)*16 + col%16)*8 + k%8].
// A B-fragment load is then one coalesced 16B/lane global_load_b128.
template <int IN, int OUT, int OUTPAD>
__global__ void expand_weights_kernel(const float* __restrict__ base_w,
                                      const float* __restrict__ spline_w,
                                      const float* __restrict__ scaler,
                                      bf16_t* __restrict__ Wexp) {
  constexpr int NCH = (IN + 31) / 32;
  constexpr int FP  = NCH * 32;     // padded feature count
  constexpr int KT8 = NCH * 36;     // K-octets per column tile (288/8 per chunk)
  int idx = blockIdx.x * 256 + threadIdx.x;
  if (idx >= OUTPAD * FP) return;
  int col = idx / FP, g = idx % FP;
  bf16_t v[9];
  if (col < OUT && g < IN) {
    size_t wi = (size_t)col * IN + g;
    float bw = base_w[wi], sc = scaler[wi];
    v[0] = tobf(bw);
    const float* sw = spline_w + wi * 8;
#pragma unroll
    for (int c = 0; c < 8; ++c) v[1 + c] = tobf(sw[c] * sc);
  } else {
    bf16_t z = tobf(0.f);
#pragma unroll
    for (int j = 0; j < 9; ++j) v[j] = z;
  }
  const int ct = col >> 4, c15 = col & 15;
  const int ch = g >> 5, lf = g & 31;
  bf16_t* base = Wexp + (size_t)ct * KT8 * 128;
#pragma unroll
  for (int j = 0; j < 9; ++j) {
    int k = ch * 288 + lf * 9 + j;
    base[((size_t)(k >> 3) << 7) + (c15 << 3) + (k & 7)] = v[j];
  }
}

// ------------- fused KAN layer as a WMMA GEMM over expanded K ---------------
// Block: 256 threads (8 waves) computes 32(rows) x 128(cols).
// Each wave owns 32 rows x 16 cols: A fragments from LDS (expanded on the
// fly), B fragments straight from L2-resident pre-expanded weights.
template <int IN, int OUT, int OUTPAD>
__global__ __launch_bounds__(256)
void kan_wmma_kernel(const float* __restrict__ zin,
                     const bf16_t* __restrict__ Wexp,
                     float* __restrict__ zout) {
  constexpr int NCH = (IN + 31) / 32;
  constexpr int KT8 = NCH * 36;
  __shared__ bf16_t Alds[32][296];   // stride: 16B aligned, LDS-bank friendly

  const int tid = threadIdx.x;
  const int rowBase = blockIdx.y * 32;
  const int colBase = blockIdx.x * 128;
  const int wave = tid >> 5, lane = tid & 31;
  const int lhalf = lane >> 4, l15 = lane & 15;
  const int klane = lhalf * 8;

  const int colTile = (colBase >> 4) + wave;               // global 16-col tile
  const bf16_t* wp = Wexp + (size_t)colTile * KT8 * 128 + (l15 << 3);

  // A-fill assignment: each thread owns one row and 4 contiguous features.
  const int arow_fill = tid >> 3;
  const int af4 = (tid & 7) * 4;
  const float* zrow = zin + (size_t)(rowBase + arow_fill) * IN;

  v8f acc0 = {}, acc1 = {};     // rows 0-15 / 16-31 of the wave tile

  for (int ch = 0; ch < NCH; ++ch) {
    __syncthreads();
    // ---- fill A: 32 rows x 32 features, one float4 load per thread ----
    {
      const int g0 = ch * 32 + af4;
      float xv[4];
      if (g0 + 3 < IN) {
        v4f xx = *(const v4f*)(zrow + g0);               // global_load_b128
        xv[0] = xx.x; xv[1] = xx.y; xv[2] = xx.z; xv[3] = xx.w;
      } else {
#pragma unroll
        for (int e = 0; e < 4; ++e)
          xv[e] = (g0 + e < IN) ? zrow[g0 + e] : 0.0f;
      }
      if (ch + 1 < NCH && g0 + 32 < IN)                  // prefetch next chunk
        __builtin_prefetch(zrow + g0 + 32, 0, 1);
#pragma unroll
      for (int e = 0; e < 4; ++e) {
        bf16_t v[9];
        if (g0 + e < IN) {
          expand9(xv[e], v);
        } else {
          bf16_t z = tobf(0.f);
#pragma unroll
          for (int j = 0; j < 9; ++j) v[j] = z;
        }
#pragma unroll
        for (int j = 0; j < 9; ++j) Alds[arow_fill][(af4 + e) * 9 + j] = v[j];
      }
    }
    __syncthreads();

    const bf16_t* arow0 = &Alds[l15][0];
    const bf16_t* arow1 = &Alds[16 + l15][0];
    const bf16_t* wchunk = wp + ((size_t)(ch * 36) << 7);
#pragma unroll
    for (int s = 0; s < 9; ++s) {
      const int kb = s * 32 + klane;
      const int k8 = (s << 2) + lhalf;               // kb / 8 within chunk
      v8bf blo = *(const v8bf*)(wchunk + ((size_t)k8 << 7));
      v8bf bhi = *(const v8bf*)(wchunk + ((size_t)(k8 + 2) << 7));
      v8bf a0l = *(const v8bf*)(arow0 + kb);         // ds_load_b128
      v8bf a0h = *(const v8bf*)(arow0 + kb + 16);
      v8bf a1l = *(const v8bf*)(arow1 + kb);
      v8bf a1h = *(const v8bf*)(arow1 + kb + 16);
      v16bf bb = __builtin_shufflevector(blo, bhi, 0,1,2,3,4,5,6,7,8,9,10,11,12,13,14,15);
      v16bf a0 = __builtin_shufflevector(a0l, a0h, 0,1,2,3,4,5,6,7,8,9,10,11,12,13,14,15);
      v16bf a1 = __builtin_shufflevector(a1l, a1h, 0,1,2,3,4,5,6,7,8,9,10,11,12,13,14,15);
      acc0 = __builtin_amdgcn_wmma_f32_16x16x32_bf16(false, a0, false, bb,
                                                     (short)0, acc0, false, false);
      acc1 = __builtin_amdgcn_wmma_f32_16x16x32_bf16(false, a1, false, bb,
                                                     (short)0, acc1, false, false);
    }
  }

  // ---- store D (lanes 0-15: M=r; lanes 16-31: M=r+8; N = lane&15) ----
  int n = colBase + wave * 16 + l15;
  if (n < OUT) {
    int m0 = rowBase + lhalf * 8;
    int m1 = rowBase + 16 + lhalf * 8;
#pragma unroll
    for (int r = 0; r < 8; ++r) {
      zout[(size_t)(m0 + r) * OUT + n] = acc0[r];
      zout[(size_t)(m1 + r) * OUT + n] = acc1[r];
    }
  }
}

// ------------- (B*C, T) -> (B,T,C) tiled transpose + residual add -----------
__global__ void add_back_kernel(const float* __restrict__ X,
                                const float* __restrict__ z2,
                                float* __restrict__ out) {
  __shared__ float tile[32][33];
  const int b = blockIdx.z;
  const int tBase = blockIdx.y * 32;
  const int cBase = blockIdx.x * 32;
  const int lane = threadIdx.x & 31, grp = threadIdx.x >> 5;
#pragma unroll
  for (int q = 0; q < 4; ++q) {
    int cl = grp + 8 * q, tl = lane;
    int t = tBase + tl, c = cBase + cl;
    float v = 0.f;
    if (t < T_DIM) v = z2[((size_t)b * C_DIM + c) * T_DIM + t];
    tile[tl][cl] = v;
  }
  __syncthreads();
#pragma unroll
  for (int q = 0; q < 4; ++q) {
    int tl = grp + 8 * q, cl = lane;
    int t = tBase + tl, c = cBase + cl;
    if (t < T_DIM) {
      size_t gi = ((size_t)b * T_DIM + t) * C_DIM + c;
      out[gi] = X[gi] + tile[tl][cl];
    }
  }
}

extern "C" void kernel_launch(void* const* d_in, const int* in_sizes, int n_in,
                              void* d_out, int out_size, void* d_ws, size_t ws_size,
                              hipStream_t stream) {
  const float* X   = (const float*)d_in[0];
  const float* lnw = (const float*)d_in[1];
  const float* lnb = (const float*)d_in[2];
  const float* bw1 = (const float*)d_in[3];
  const float* sw1 = (const float*)d_in[4];
  const float* sc1 = (const float*)d_in[5];
  const float* bw2 = (const float*)d_in[6];
  const float* sw2 = (const float*)d_in[7];
  const float* sc2 = (const float*)d_in[8];
  float* out = (float*)d_out;

  // layer 1: IN=196 (NCH=7, K'=2016), OUT=384, OUTPAD=384 (3 x 128-col blocks)
  // layer 2: IN=384 (NCH=12, K'=3456), OUT=196, OUTPAD=256 (2 x 128-col blocks)
  constexpr size_t WEXP1_BYTES = (size_t)384 * 7 * 288 * 2;   // 1,548,288
  constexpr size_t WEXP2_BYTES = (size_t)256 * 12 * 288 * 2;  // 1,769,472

  char*   ws      = (char*)d_ws;
  float*  stats   = (float*)ws;                         // 512 B
  float*  partial = (float*)(ws + 1024);                // 64*128*2 f32 = 64 KB
  bf16_t* wexp1   = (bf16_t*)(ws + 1024 + 65536);
  bf16_t* wexp2   = (bf16_t*)(ws + 1024 + 65536 + WEXP1_BYTES);
  float*  z0      = (float*)(ws + 1024 + 65536 + WEXP1_BYTES + WEXP2_BYTES);
  float*  z1      = z0 + (size_t)NROWS * T_DIM;
  // layer-2 output aliases z0 (layer 2 reads only z1)

  // weight expansion (once per call; tiny)
  {
    int n1 = 384 * (7 * 32);
    expand_weights_kernel<T_DIM, H_DIM, 384>
        <<<(n1 + 255) / 256, 256, 0, stream>>>(bw1, sw1, sc1, wexp1);
    int n2 = 256 * (12 * 32);
    expand_weights_kernel<H_DIM, T_DIM, 256>
        <<<(n2 + 255) / 256, 256, 0, stream>>>(bw2, sw2, sc2, wexp2);
  }

  ln_partial_kernel<<<dim3(128, B_DIM), 256, 0, stream>>>(X, partial);
  ln_final_kernel<<<B_DIM, 128, 0, stream>>>(partial, stats);

  dim3 tgrid(C_DIM / 32, (T_DIM + 31) / 32, B_DIM);
  norm_transpose_kernel<<<tgrid, 256, 0, stream>>>(X, lnw, lnb, stats, z0);

  kan_wmma_kernel<T_DIM, H_DIM, 384>
      <<<dim3(3, NROWS / 32), 256, 0, stream>>>(z0, wexp1, z1);
  kan_wmma_kernel<H_DIM, T_DIM, 256>
      <<<dim3(2, NROWS / 32), 256, 0, stream>>>(z1, wexp2, z0);

  add_back_kernel<<<tgrid, 256, 0, stream>>>(X, z0, out);
}
